// GNN_91173565759520
// MI455X (gfx1250) — matrix operand order repaired and checked by reading, not compiled, not run on hardware
//
#include <hip/hip_runtime.h>
#include <math.h>

#define HD 128
#define DINC 64
#define XPITCH 40   // halves per A-tile row: 80B, 16B-aligned, conflict-free
#define ROWS_PER_BLOCK 128

typedef __attribute__((ext_vector_type(16))) _Float16 v16h;
typedef __attribute__((ext_vector_type(8)))  _Float16 v8h;
typedef __attribute__((ext_vector_type(8)))  float    v8f;

// 1/sqrt(1 + 1e-5)
#define BN_INV 0.99999500003750f

union HPack { _Float16 h[2]; unsigned u; };

// ---------------- threefry2x32 (exact JAX) ----------------
__device__ __forceinline__ unsigned rotl32(unsigned v, int d) {
  return (v << d) | (v >> (32 - d));
}

__device__ __forceinline__ void tf2x32(unsigned k0, unsigned k1,
                                       unsigned c0, unsigned c1,
                                       unsigned& o0, unsigned& o1) {
  unsigned ks2 = k0 ^ k1 ^ 0x1BD11BDAu;
  unsigned x0 = c0 + k0, x1 = c1 + k1;
#define RND(r) { x0 += x1; x1 = rotl32(x1, r); x1 ^= x0; }
  RND(13) RND(15) RND(26) RND(6)   x0 += k1;  x1 += ks2 + 1u;
  RND(17) RND(29) RND(16) RND(24)  x0 += ks2; x1 += k0  + 2u;
  RND(13) RND(15) RND(26) RND(6)   x0 += k0;  x1 += k1  + 3u;
  RND(17) RND(29) RND(16) RND(24)  x0 += k1;  x1 += ks2 + 4u;
  RND(13) RND(15) RND(26) RND(6)   x0 += ks2; x1 += k0  + 5u;
#undef RND
  o0 = x0; o1 = x1;
}

__device__ __forceinline__ float bits_to_u01(unsigned bits) {
  unsigned u = (bits >> 9) | 0x3f800000u;
  float f = __uint_as_float(u) - 1.0f;
  f = f * (1.0f - 1e-6f) + 1e-6f;
  return fmaxf(f, 1e-6f);
}

// ---------------- zero ----------------
__global__ void k_zero(float* __restrict__ p, int n) {
  int i = blockIdx.x * blockDim.x + threadIdx.x;
  if (i < n) p[i] = 0.0f;
}

// ---------------- WMMA GEMM: out[N x 128] = epi(A[N x K] @ W[K x 128]) ----------------
// Block: 256 threads (8 waves), 128-row x 128-col output tile.
// W packed in LDS per-lane-contiguous: [kblock][tile][half*16+n][8 dwords], so a
// B fragment is 2x ds_load_b128 with immediate offsets off one base (16-bit DS
// offset covers all 8 tiles). A tile rows pitch 40 halves -> A fragment is also
// 2x ds_load_b128. Per K-step: 18 DS ops, 8 back-to-back v_wmma, ~no addr VALU.
template <int K>
__global__ void __launch_bounds__(256)
k_gemm_wmma(const float* __restrict__ A, const float* __restrict__ W,
            const float* __restrict__ bias, const float* __restrict__ gamma,
            const float* __restrict__ beta, const float* res,
            float* out, int N, int relu) {
  __shared__ unsigned  wlds[(K / 32) * 2048];          // packed B fragments
  __shared__ _Float16  xlds[ROWS_PER_BLOCK * XPITCH];  // A tile: [row][k%32]

  const int tid  = threadIdx.x;
  const int lane = tid & 31;
  const int wave = tid >> 5;
  const int half = lane >> 4;
  const int l15  = lane & 15;
  const int rowbase = blockIdx.x * ROWS_PER_BLOCK;

  // stage W (K x 128, row-major) into packed fragment layout, f16 pairs
  for (int i = tid; i < (K / 2) * HD; i += 256) {
    int col = i & 127, kp = i >> 7;
    int k = kp << 1;
    HPack p;
    p.h[0] = (_Float16)W[(size_t)k * HD + col];
    p.h[1] = (_Float16)W[(size_t)(k + 1) * HD + col];
    int t = col >> 4, n = col & 15;
    int hb = (k >> 3) & 1;
    int slot = ((k >> 5) << 11) + (t << 8) + (((hb << 4) + n) << 3) + (((k >> 4) & 1) << 2) + (kp & 3);
    wlds[slot] = p.u;
  }

  v8f acc[8];
  v8f zz = {};
#pragma unroll
  for (int t = 0; t < 8; ++t) acc[t] = zz;

  __syncthreads();

#pragma unroll 1
  for (int ks = 0; ks < K; ks += 32) {
    // stage ROWS_PER_BLOCK x 32 A tile as f16 (+ prefetch next tile)
    for (int i = tid; i < ROWS_PER_BLOCK * 32; i += 256) {
      int r = i >> 5, c = i & 31;
      int row = rowbase + r;
      float v = (row < N) ? A[(size_t)row * K + ks + c] : 0.0f;
      xlds[r * XPITCH + c] = (_Float16)v;
      if (ks + 32 < K && row < N)
        __builtin_prefetch(&A[(size_t)row * K + ks + 32 + c], 0, 3);
    }
    __syncthreads();

    // A fragment: element pairs (2v,2v+1) need k = (v>>2)*16 + half*8 + (v&3)*2
    // -> two contiguous 16B runs: [half*8 .. +7] and [16+half*8 .. +7]
    const _Float16* ap = &xlds[(wave * 16 + l15) * XPITCH];
    v8h alo = *(const v8h*)(ap + (half << 3));
    v8h ahi = *(const v8h*)(ap + 16 + (half << 3));
    v16h afr = __builtin_shufflevector(alo, ahi, 0, 1, 2, 3, 4, 5, 6, 7,
                                       8, 9, 10, 11, 12, 13, 14, 15);

    // B fragments: per-lane-contiguous 8 dwords; tile stride 256 dwords
    const unsigned* bb = &wlds[((ks >> 5) << 11) + (((half << 4) + l15) << 3)];
    v16h bfr[8];
#pragma unroll
    for (int t = 0; t < 8; ++t) {
      v8h blo = *(const v8h*)(bb + t * 256);
      v8h bhi = *(const v8h*)(bb + t * 256 + 4);
      bfr[t] = __builtin_shufflevector(blo, bhi, 0, 1, 2, 3, 4, 5, 6, 7,
                                       8, 9, 10, 11, 12, 13, 14, 15);
    }

    // 8 back-to-back WMMAs; distinct destinations, shared sources -> no hazards
#pragma unroll
    for (int t = 0; t < 8; ++t) {
      acc[t] = __builtin_amdgcn_wmma_f32_16x16x32_f16(
          false, afr, false, bfr[t], (short)0, acc[t], false, false);
    }
    __syncthreads();
  }

  // epilogue; C/D layout: VGPR r -> M = half*8 + r, N = lane&15
#pragma unroll
  for (int t = 0; t < 8; ++t) {
    const int col = t * 16 + l15;
    const float b  = bias[col];
    const float g  = gamma ? gamma[col] : 0.0f;
    const float be = beta ? beta[col] : 0.0f;
#pragma unroll
    for (int r = 0; r < 8; ++r) {
      const int row = rowbase + wave * 16 + (half << 3) + r;
      if (row < N) {
        float v = acc[t][r] + b;
        if (gamma) v = g * v * BN_INV + be;
        if (relu)  v = fmaxf(v, 0.0f);
        if (res)   v += res[(size_t)row * HD + col];
        out[(size_t)row * HD + col] = v;
      }
    }
  }
}

// ---------------- per-node edge-gate projections: ps0,ps1,pd0,pd1 ----------------
__global__ void k_proj(const float* __restrict__ x, const float* __restrict__ wec,
                       float* __restrict__ proj, int N) {
  int wid  = (blockIdx.x * blockDim.x + threadIdx.x) >> 5;
  int lane = threadIdx.x & 31;
  if (wid >= N) return;
  float a0 = 0.f, a1 = 0.f, a2 = 0.f, a3 = 0.f;
  for (int k = lane; k < HD; k += 32) {
    float xv = x[(size_t)wid * HD + k];
    a0 += xv * wec[k * 2 + 0];
    a1 += xv * wec[k * 2 + 1];
    a2 += xv * wec[(HD + k) * 2 + 0];
    a3 += xv * wec[(HD + k) * 2 + 1];
  }
#pragma unroll
  for (int off = 16; off > 0; off >>= 1) {
    a0 += __shfl_down(a0, off);
    a1 += __shfl_down(a1, off);
    a2 += __shfl_down(a2, off);
    a3 += __shfl_down(a3, off);
  }
  if (lane == 0) {
    proj[wid * 4 + 0] = a0;
    proj[wid * 4 + 1] = a1;
    proj[wid * 4 + 2] = a2;
    proj[wid * 4 + 3] = a3;
  }
}

// ---------------- edge gate + gated scatter-add (one wave per edge) ----------------
__global__ void k_edge(const float* __restrict__ x, const int* __restrict__ ei,
                       const float* __restrict__ ea, const float* __restrict__ proj,
                       const float* __restrict__ wec, const float* __restrict__ bec,
                       float* __restrict__ agg, int layer, int E) {
  int e = (blockIdx.x * blockDim.x + threadIdx.x) >> 5;
  if (e >= E) return;
  int lane = threadIdx.x & 31;
  int src = ei[e];
  int dst = ei[E + e];
  float a = ea[e];

  // gate logits (wave-uniform; redundant lanes are free under SIMD)
  float z0 = proj[src * 4 + 0] + proj[dst * 4 + 2] + a * wec[256 * 2 + 0] + bec[0];
  float z1 = proj[src * 4 + 1] + proj[dst * 4 + 3] + a * wec[256 * 2 + 1] + bec[1];
  float s0 = 1.0f / (1.0f + __expf(-z0));
  float s1 = 1.0f / (1.0f + __expf(-z1));

  // folded key: threefry(key(42), (0, layer))
  unsigned fk0, fk1;
  tf2x32(0u, 42u, 0u, (unsigned)layer, fk0, fk1);

  // JAX counter-mode bits for flat indices 2e, 2e+1 over n = 2E (half = E)
  unsigned uE = (unsigned)E;
  unsigned i0 = 2u * (unsigned)e, i1 = i0 + 1u;
  unsigned o0, o1, b0, b1;
  if (i0 < uE) { tf2x32(fk0, fk1, i0, i0 + uE, o0, o1); b0 = o0; }
  else         { tf2x32(fk0, fk1, i0 - uE, i0, o0, o1); b0 = o1; }
  if (i1 < uE) { tf2x32(fk0, fk1, i1, i1 + uE, o0, o1); b1 = o0; }
  else         { tf2x32(fk0, fk1, i1 - uE, i1, o0, o1); b1 = o1; }

  float g0 = -logf(-logf(bits_to_u01(b0)));
  float g1 = -logf(-logf(bits_to_u01(b1)));

  // forward value of straight-through Gumbel: ew = 1 iff argmax == 0
  if (s0 + g0 >= s1 + g1) {
    const float* xs = x + (size_t)src * HD;
    float* ad = agg + (size_t)dst * HD;
#pragma unroll
    for (int j = 0; j < 4; ++j) {
      int c = lane + 32 * j;
      (void)__hip_atomic_fetch_add(&ad[c], xs[c], __ATOMIC_RELAXED,
                                   __HIP_MEMORY_SCOPE_AGENT);
    }
  }
}

// ---------------- final GEMV: out[n] = x[n,:] @ W4 + b4 ----------------
__global__ void k_final(const float* __restrict__ x, const float* __restrict__ W4,
                        const float* __restrict__ b4, float* __restrict__ out, int N) {
  int wid  = (blockIdx.x * blockDim.x + threadIdx.x) >> 5;
  int lane = threadIdx.x & 31;
  if (wid >= N) return;
  float acc = 0.f;
  for (int k = lane; k < HD; k += 32) acc += x[(size_t)wid * HD + k] * W4[k];
#pragma unroll
  for (int off = 16; off > 0; off >>= 1) acc += __shfl_down(acc, off);
  if (lane == 0) out[wid] = acc + b4[0];
}

extern "C" void kernel_launch(void* const* d_in, const int* in_sizes, int n_in,
                              void* d_out, int out_size, void* d_ws, size_t ws_size,
                              hipStream_t stream) {
  (void)n_in; (void)out_size; (void)ws_size;
  const float* x    = (const float*)d_in[0];
  const int*   ei   = (const int*)d_in[1];
  const float* ea   = (const float*)d_in[2];
  const float* preW = (const float*)d_in[3];
  const float* preb = (const float*)d_in[4];
  const float* Wec  = (const float*)d_in[5];
  const float* bec  = (const float*)d_in[6];
  const float* Wm   = (const float*)d_in[7];
  const float* bm   = (const float*)d_in[8];
  const float* gm   = (const float*)d_in[9];
  const float* bnm  = (const float*)d_in[10];
  const float* W1   = (const float*)d_in[11];
  const float* b1   = (const float*)d_in[12];
  const float* W2   = (const float*)d_in[13];
  const float* b2   = (const float*)d_in[14];
  const float* W3   = (const float*)d_in[15];
  const float* b3   = (const float*)d_in[16];
  const float* W4   = (const float*)d_in[17];
  const float* b4   = (const float*)d_in[18];
  const float* gBN  = (const float*)d_in[19];
  const float* bBN  = (const float*)d_in[20];
  float* out = (float*)d_out;

  const int N = in_sizes[0] / DINC;     // 50000
  const int E = in_sizes[1] / 2;        // 800000
  const size_t NH = (size_t)N * HD;

  float* ws   = (float*)d_ws;
  float* X    = ws;                     // current node features  [N,128]
  float* AGG  = ws + NH;                // aggregation buffer     [N,128]
  float* Hb   = ws + 2 * NH;            // hidden buffer          [N,128]
  float* proj = ws + 3 * NH;            // per-node gate proj     [N,4]

  const int gemm_grid = (N + ROWS_PER_BLOCK - 1) / ROWS_PER_BLOCK;
  const int wave_grid = (N + 7) / 8;        // 1 wave per node, 8 waves/block
  const int edge_grid = (E + 7) / 8;        // 1 wave per edge
  const int zero_n    = N * HD;
  const int zero_grid = (zero_n + 255) / 256;

  // pre-layer: X = relu(x @ preW + preb)
  k_gemm_wmma<DINC><<<gemm_grid, 256, 0, stream>>>(x, preW, preb, nullptr, nullptr,
                                                   nullptr, X, N, 1);

  for (int l = 0; l < 3; ++l) {
    const float* wecL = Wec + (size_t)l * 257 * 2;
    const float* becL = bec + (size_t)l * 2;

    k_proj<<<wave_grid, 256, 0, stream>>>(X, wecL, proj, N);
    k_zero<<<zero_grid, 256, 0, stream>>>(AGG, zero_n);
    k_edge<<<edge_grid, 256, 0, stream>>>(X, ei, ea, proj, wecL, becL, AGG, l, E);

    // h = relu(bn(agg @ Wm[l,0] + bm[l,0]))
    k_gemm_wmma<HD><<<gemm_grid, 256, 0, stream>>>(
        AGG, Wm + (size_t)l * 2 * HD * HD, bm + (size_t)l * 2 * HD,
        gm + (size_t)l * 2 * HD, bnm + (size_t)l * 2 * HD, nullptr, Hb, N, 1);
    // X = relu(bn(h @ Wm[l,1] + bm[l,1])) + X   (res aliases out: safe per-thread)
    k_gemm_wmma<HD><<<gemm_grid, 256, 0, stream>>>(
        Hb, Wm + ((size_t)l * 2 + 1) * HD * HD, bm + ((size_t)l * 2 + 1) * HD,
        gm + ((size_t)l * 2 + 1) * HD, bnm + ((size_t)l * 2 + 1) * HD, X, X, N, 1);
  }

  // output MLP
  k_gemm_wmma<HD><<<gemm_grid, 256, 0, stream>>>(X,  W1, b1, gBN, bBN, nullptr, AGG, N, 1);
  k_gemm_wmma<HD><<<gemm_grid, 256, 0, stream>>>(AGG, W2, b2, gBN, bBN, nullptr, Hb,  N, 1);
  k_gemm_wmma<HD><<<gemm_grid, 256, 0, stream>>>(Hb,  W3, b3, gBN, bBN, nullptr, AGG, N, 1);
  k_final<<<wave_grid, 256, 0, stream>>>(AGG, W4, b4, out, N);
}
